// DiscreteDistributionBlock1D_23613730194017
// MI455X (gfx1250) — matrix operand order repaired and checked by reading, not compiled, not run on hardware
//
#include <hip/hip_runtime.h>
#include <hip/hip_bf16.h>

typedef __attribute__((ext_vector_type(16))) __bf16 bf16x16;
typedef __attribute__((ext_vector_type(8)))  float  v8f;

#define LSEQ 1024
#define NB   16

__device__ __forceinline__ v8f wmma_bf16(bf16x16 a, bf16x16 b, v8f c) {
  return __builtin_amdgcn_wmma_f32_16x16x32_bf16(false, a, false, b,
                                                 (short)0, c, false, false);
}

// ---------------------------------------------------------------------------
// Fused GroupNorm(32 groups) + SiLU, writing bf16 in transposed (b, t, c)
// layout so conv B-fragments are contiguous-32B loads.
// grid = NB*32 blocks, 256 threads.
// ---------------------------------------------------------------------------
__global__ void gn_silu_kernel(const float* __restrict__ x,
                               const float* __restrict__ w,
                               const float* __restrict__ bias,
                               __bf16* __restrict__ outT,
                               int C, int cpg) {
  const int b = blockIdx.x >> 5;
  const int g = blockIdx.x & 31;
  const int n = cpg * LSEQ;
  const float* xg = x + ((size_t)b * C + (size_t)g * cpg) * LSEQ;

  float s = 0.f, s2 = 0.f;
  for (int i = threadIdx.x; i < n; i += 256) {
    float v = xg[i];
    s += v; s2 += v * v;
  }
  __shared__ float sh[512];
  sh[threadIdx.x] = s;
  sh[256 + threadIdx.x] = s2;
  __syncthreads();
  for (int o = 128; o > 0; o >>= 1) {
    if (threadIdx.x < o) {
      sh[threadIdx.x] += sh[threadIdx.x + o];
      sh[256 + threadIdx.x] += sh[256 + threadIdx.x + o];
    }
    __syncthreads();
  }
  const float mean = sh[0] / (float)n;
  const float var  = sh[256] / (float)n - mean * mean;
  const float rs   = rsqrtf(var + 1e-5f);

  for (int i = threadIdx.x; i < n; i += 256) {
    int c = g * cpg + i / LSEQ;
    int t = i % LSEQ;
    float v = (xg[i] - mean) * rs * w[c] + bias[c];
    v = v / (1.0f + __expf(-v));            // SiLU
    outT[((size_t)b * LSEQ + t) * C + c] = (__bf16)v;
  }
}

// fp32 (b,c,t) -> bf16 transposed (b,t,c)
__global__ void to_bf16T_kernel(const float* __restrict__ x,
                                __bf16* __restrict__ outT, int C) {
  size_t idx = (size_t)blockIdx.x * 256 + threadIdx.x;
  size_t total = (size_t)NB * C * LSEQ;
  if (idx >= total) return;
  int t = (int)(idx % LSEQ);
  size_t r = idx / LSEQ;
  int c = (int)(r % C);
  int b = (int)(r / C);
  outT[((size_t)b * LSEQ + t) * C + c] = (__bf16)x[idx];
}

// ---------------------------------------------------------------------------
// Repack conv weights (Cout, Cin, KW) fp32 into WMMA A-fragment order:
// frag[mt][dk][cc][lane][e] bf16, contiguous k order matching B loads:
//   m = mt*16 + (lane&15); k = 16*(lane>>4) + e; ci = cc*32+k
// ---------------------------------------------------------------------------
__global__ void repack_w_kernel(const float* __restrict__ w,
                                __bf16* __restrict__ frag,
                                int Cout, int Cin, int KW) {
  size_t idx = (size_t)blockIdx.x * 256 + threadIdx.x;
  size_t total = (size_t)Cout * Cin * KW;
  if (idx >= total) return;
  const int nCC = Cin >> 5;
  int e = (int)(idx & 15);
  size_t r = idx >> 4;
  int lane = (int)(r & 31); r >>= 5;
  int cc = (int)(r % nCC); r /= nCC;
  int dk = (int)(r % KW);  r /= KW;
  int mt = (int)r;
  int m  = mt * 16 + (lane & 15);
  int h  = lane >> 4;
  int kk = 16 * h + e;
  int ci = cc * 32 + kk;
  frag[idx] = (__bf16)w[((size_t)m * Cin + ci) * KW + dk];
}

// ---------------------------------------------------------------------------
// Implicit-GEMM conv1d via WMMA bf16->f32, compile-time shapes.
// Each wave: 32 co (2 M-tiles) x 64 t (4 N-tiles) = 64 acc VGPRs; B fragment
// loads shared across both M-tiles.  Block = 8 waves = 256 co.
// grid = NB * (LSEQ/64), 256 threads.  Fused bias + optional residual.
// ---------------------------------------------------------------------------
template <int CIN, int COUT, int KW>
__launch_bounds__(256)
__global__ void conv_wmma_kernel(const __bf16* __restrict__ xT,
                                 const __bf16* __restrict__ wfrag,
                                 const float* __restrict__ bias,
                                 const float* __restrict__ residual,
                                 float* __restrict__ out) {
  constexpr int PAD = (KW == 3) ? 1 : 0;
  constexpr int NCC = CIN / 32;
  const int tg = blockIdx.x & (LSEQ / 64 - 1);
  const int b  = blockIdx.x >> 4;            // LSEQ/64 == 16
  const int wave = threadIdx.x >> 5;
  const int lane = threadIdx.x & 31;
  const int mt0  = wave * 2;                 // 8 waves cover COUT=256
  const int t0   = tg * 64;
  const int ln = lane & 15, h = lane >> 4;

  v8f acc[2][4];
  #pragma unroll
  for (int mi = 0; mi < 2; ++mi)
    #pragma unroll
    for (int j = 0; j < 4; ++j) acc[mi][j] = (v8f){};

  const __bf16* xb = xT + (size_t)b * LSEQ * CIN;

  #pragma unroll
  for (int dk = 0; dk < KW; ++dk) {
    #pragma unroll
    for (int cc = 0; cc < NCC; ++cc) {
      bf16x16 a0 = *(const bf16x16*)(wfrag +
          ((((size_t)mt0 * KW + dk) * NCC + cc) << 9) + ((size_t)lane << 4));
      bf16x16 a1 = *(const bf16x16*)(wfrag +
          ((((size_t)(mt0 + 1) * KW + dk) * NCC + cc) << 9) + ((size_t)lane << 4));
      const int cib = (cc << 5) + (h << 4);
      #pragma unroll
      for (int j = 0; j < 4; ++j) {
        int t = t0 + j * 16 + ln + dk - PAD;
        bf16x16 bv = {};
        if (t >= 0 && t < LSEQ)
          bv = *(const bf16x16*)(xb + (size_t)t * CIN + cib);
        acc[0][j] = wmma_bf16(a0, bv, acc[0][j]);
        acc[1][j] = wmma_bf16(a1, bv, acc[1][j]);
      }
    }
  }

  #pragma unroll
  for (int mi = 0; mi < 2; ++mi) {
    #pragma unroll
    for (int j = 0; j < 4; ++j) {
      int t = t0 + j * 16 + ln;
      #pragma unroll
      for (int v = 0; v < 8; ++v) {
        int co = (mt0 + mi) * 16 + v + 8 * h;
        size_t oi = ((size_t)b * COUT + co) * LSEQ + t;
        float val = acc[mi][j][v] + bias[co];
        if (residual) val += residual[oi];
        out[oi] = val;
      }
    }
  }
}

// ---------------------------------------------------------------------------
// Head GEMM (per (b,k)) + fused squared-error reduction vs target.
// grid = NB*64 blocks, 256 threads (8 waves x 128 t each).
// B fragments of each N-tile cached in registers, reused across 5 M-tiles.
// dist[b*64+k] = mean over (80 mel, 1024 t) of (pred - target)^2.
// ---------------------------------------------------------------------------
__launch_bounds__(256)
__global__ void head_dist_kernel(const __bf16* __restrict__ finT,
                                 const __bf16* __restrict__ wfrag,
                                 const float* __restrict__ hb,
                                 const float* __restrict__ target,
                                 float* __restrict__ dist) {
  const int b = blockIdx.x >> 6;
  const int k = blockIdx.x & 63;
  const int wave = threadIdx.x >> 5, lane = threadIdx.x & 31;
  const int ln = lane & 15, h = lane >> 4;
  const __bf16* xb = finT + (size_t)b * LSEQ * 256;

  float psum = 0.f;
  for (int j = 0; j < 8; ++j) {
    const int t0 = wave * 128 + j * 16;
    bf16x16 bv[8];
    #pragma unroll
    for (int cc = 0; cc < 8; ++cc)
      bv[cc] = *(const bf16x16*)(xb + (size_t)(t0 + ln) * 256 + (cc << 5) + (h << 4));
    #pragma unroll
    for (int mm = 0; mm < 5; ++mm) {
      const int mt = k * 5 + mm;
      v8f acc = {};
      #pragma unroll
      for (int cc = 0; cc < 8; ++cc) {
        bf16x16 a = *(const bf16x16*)(wfrag +
            (((size_t)mt * 8 + cc) << 9) + ((size_t)lane << 4));
        acc = wmma_bf16(a, bv[cc], acc);
      }
      const int t = t0 + ln;
      #pragma unroll
      for (int v = 0; v < 8; ++v) {
        int mel = mm * 16 + v + 8 * h;
        float pred = acc[v] + hb[k * 80 + mel];
        float d = pred - target[((size_t)b * 80 + mel) * LSEQ + t];
        psum += d * d;
      }
    }
  }
  __shared__ float sh[256];
  sh[threadIdx.x] = psum;
  __syncthreads();
  for (int o = 128; o > 0; o >>= 1) {
    if (threadIdx.x < o) sh[threadIdx.x] += sh[threadIdx.x + o];
    __syncthreads();
  }
  if (threadIdx.x == 0)
    dist[b * 64 + k] = sh[0] / (80.0f * (float)LSEQ);
}

// argmin over k per batch + mean-of-min loss.  1 block, 32 threads.
__global__ void argmin_kernel(const float* __restrict__ dist,
                              int* __restrict__ bestIdx,
                              float* __restrict__ loss) {
  __shared__ float mins[16];
  if (threadIdx.x < 16) {
    int b = threadIdx.x;
    float mv = dist[b * 64]; int mi = 0;
    for (int kk = 1; kk < 64; ++kk) {
      float v = dist[b * 64 + kk];
      if (v < mv) { mv = v; mi = kk; }
    }
    bestIdx[b] = mi;
    mins[b] = mv;
  }
  __syncthreads();
  if (threadIdx.x == 0) {
    float s = 0.f;
    for (int b = 0; b < 16; ++b) s += mins[b];
    *loss = s / 16.0f;
  }
}

// Recompute only the winning codebook slice: best (b, 80, 1024).
// grid = NB*8 blocks (8 t-groups of 128), 256 threads.
__launch_bounds__(256)
__global__ void head_best_kernel(const __bf16* __restrict__ finT,
                                 const __bf16* __restrict__ wfrag,
                                 const float* __restrict__ hb,
                                 const int* __restrict__ bestIdx,
                                 float* __restrict__ best) {
  const int b  = blockIdx.x >> 3;
  const int tg = blockIdx.x & 7;
  const int wave = threadIdx.x >> 5, lane = threadIdx.x & 31;
  const int ln = lane & 15, h = lane >> 4;
  const int k = bestIdx[b];
  const int t0 = tg * 128 + wave * 16;
  const __bf16* xb = finT + (size_t)b * LSEQ * 256;

  bf16x16 bv[8];
  #pragma unroll
  for (int cc = 0; cc < 8; ++cc)
    bv[cc] = *(const bf16x16*)(xb + (size_t)(t0 + ln) * 256 + (cc << 5) + (h << 4));

  #pragma unroll
  for (int mm = 0; mm < 5; ++mm) {
    const int mt = k * 5 + mm;
    v8f acc = {};
    #pragma unroll
    for (int cc = 0; cc < 8; ++cc) {
      bf16x16 a = *(const bf16x16*)(wfrag +
          (((size_t)mt * 8 + cc) << 9) + ((size_t)lane << 4));
      acc = wmma_bf16(a, bv[cc], acc);
    }
    const int t = t0 + ln;
    #pragma unroll
    for (int v = 0; v < 8; ++v) {
      int mel = mm * 16 + v + 8 * h;
      best[((size_t)b * 80 + mel) * LSEQ + t] = acc[v] + hb[k * 80 + mel];
    }
  }
}

// ---------------------------------------------------------------------------
extern "C" void kernel_launch(void* const* d_in, const int* in_sizes, int n_in,
                              void* d_out, int out_size, void* d_ws, size_t ws_size,
                              hipStream_t stream) {
  (void)in_sizes; (void)n_in; (void)out_size; (void)ws_size;
  const float* x        = (const float*)d_in[0];
  const float* target   = (const float*)d_in[1];
  const float* b0_n0w   = (const float*)d_in[2];
  const float* b0_n0b   = (const float*)d_in[3];
  const float* b0_c0w   = (const float*)d_in[4];
  const float* b0_c0b   = (const float*)d_in[5];
  const float* b0_n1w   = (const float*)d_in[6];
  const float* b0_n1b   = (const float*)d_in[7];
  const float* b0_c1w   = (const float*)d_in[8];
  const float* b0_c1b   = (const float*)d_in[9];
  const float* b0_skw   = (const float*)d_in[10];
  const float* b0_skb   = (const float*)d_in[11];
  const float* bs_n0w   = (const float*)d_in[12];
  const float* bs_n0b   = (const float*)d_in[13];
  const float* bs_c0w   = (const float*)d_in[14];
  const float* bs_c0b   = (const float*)d_in[15];
  const float* bs_n1w   = (const float*)d_in[16];
  const float* bs_n1b   = (const float*)d_in[17];
  const float* bs_c1w   = (const float*)d_in[18];
  const float* bs_c1b   = (const float*)d_in[19];
  const float* head_w   = (const float*)d_in[20];
  const float* head_b   = (const float*)d_in[21];

  char* ws = (char*)d_ws;
  size_t off = 0;
  auto alloc = [&](size_t bytes) -> void* {
    void* p = ws + off;
    off = (off + bytes + 255) & ~(size_t)255;
    return p;
  };
  __bf16* xT0  = (__bf16*)alloc((size_t)NB * LSEQ * 128 * 2);
  __bf16* nrm  = (__bf16*)alloc((size_t)NB * LSEQ * 256 * 2);
  float*  P0   = (float*) alloc((size_t)NB * 256 * LSEQ * 4);
  float*  P1   = (float*) alloc((size_t)NB * 256 * LSEQ * 4);
  float*  P2   = (float*) alloc((size_t)NB * 256 * LSEQ * 4);
  __bf16* finT = (__bf16*)alloc((size_t)NB * LSEQ * 256 * 2);
  __bf16* wfrag= (__bf16*)alloc((size_t)5120 * 256 * 2);
  int*    bIdx = (int*)   alloc(64);

  float* best = (float*)d_out;
  float* dist = best + (size_t)NB * 80 * LSEQ;
  float* loss = dist + NB * 64;

  const dim3 B256(256);
  const int gnGrid   = NB * 32;
  const int convGrid = NB * (LSEQ / 64);       // 256 blocks, 8 waves each

  // ---- block 0 ----
  to_bf16T_kernel<<<(NB*128*LSEQ + 255)/256, B256, 0, stream>>>(x, xT0, 128);
  gn_silu_kernel<<<gnGrid, B256, 0, stream>>>(x, b0_n0w, b0_n0b, nrm, 128, 4);
  repack_w_kernel<<<(256*128*3)/256, B256, 0, stream>>>(b0_c0w, wfrag, 256, 128, 3);
  conv_wmma_kernel<128,256,3><<<convGrid, B256, 0, stream>>>(nrm, wfrag, b0_c0b, nullptr, P0);
  gn_silu_kernel<<<gnGrid, B256, 0, stream>>>(P0, b0_n1w, b0_n1b, nrm, 256, 8);
  repack_w_kernel<<<(256*128)/256, B256, 0, stream>>>(b0_skw, wfrag, 256, 128, 1);
  conv_wmma_kernel<128,256,1><<<convGrid, B256, 0, stream>>>(xT0, wfrag, b0_skb, nullptr, P1);
  repack_w_kernel<<<(256*256*3)/256, B256, 0, stream>>>(b0_c1w, wfrag, 256, 256, 3);
  conv_wmma_kernel<256,256,3><<<convGrid, B256, 0, stream>>>(nrm, wfrag, b0_c1b, P1, P2);

  // ---- blocks 1..3 (identity skip) ----
  float* X = P2; float* H = P0; float* O = P1;
  for (int i = 0; i < 3; ++i) {
    gn_silu_kernel<<<gnGrid, B256, 0, stream>>>(X, bs_n0w + i*256, bs_n0b + i*256, nrm, 256, 8);
    repack_w_kernel<<<(256*256*3)/256, B256, 0, stream>>>(bs_c0w + (size_t)i*256*256*3, wfrag, 256, 256, 3);
    conv_wmma_kernel<256,256,3><<<convGrid, B256, 0, stream>>>(nrm, wfrag, bs_c0b + i*256, nullptr, H);
    gn_silu_kernel<<<gnGrid, B256, 0, stream>>>(H, bs_n1w + i*256, bs_n1b + i*256, nrm, 256, 8);
    repack_w_kernel<<<(256*256*3)/256, B256, 0, stream>>>(bs_c1w + (size_t)i*256*256*3, wfrag, 256, 256, 3);
    conv_wmma_kernel<256,256,3><<<convGrid, B256, 0, stream>>>(nrm, wfrag, bs_c1b + i*256, X, O);
    float* tmp = X; X = O; O = tmp;   // next block input = this output
  }

  // ---- head: distances, argmin, winning slice ----
  to_bf16T_kernel<<<(NB*256*LSEQ + 255)/256, B256, 0, stream>>>(X, finT, 256);
  repack_w_kernel<<<(5120*256)/256, B256, 0, stream>>>(head_w, wfrag, 5120, 256, 1);
  head_dist_kernel<<<NB * 64, B256, 0, stream>>>(finT, wfrag, head_b, target, dist);
  argmin_kernel<<<1, 32, 0, stream>>>(dist, bIdx, loss);
  head_best_kernel<<<NB * 8, B256, 0, stream>>>(finT, wfrag, head_b, bIdx, best);
}
